// SparseGraphAttentionBlock_61976378081703
// MI455X (gfx1250) — compile-verified, matrix operand order checked by reading, not run on hardware
//
#include <hip/hip_runtime.h>
#include <hip/hip_bf16.h>
#include <math.h>

#define BB 2
#define NN 2048
#define DD 512
#define HH 8
#define HDIM 64
#define RR 64
#define MM (BB*NN)          /* 4096 rows */
#define ASCALE 0.125f       /* 64^-0.5 */
#define LN_EPS 1e-5f

typedef __attribute__((ext_vector_type(16))) __bf16 v16bf;
typedef __attribute__((ext_vector_type(8)))  float  v8f;
typedef __attribute__((ext_vector_type(4)))  unsigned int v4u;
typedef __attribute__((ext_vector_type(8)))  int v8i;
typedef __attribute__((ext_vector_type(4)))  int v4i;

#if defined(__has_builtin)
#if __has_builtin(__builtin_amdgcn_tensor_load_to_lds) && \
    __has_builtin(__builtin_amdgcn_s_wait_tensorcnt)
#define HAVE_TDM 1
#endif
#endif
#ifndef HAVE_TDM
#define HAVE_TDM 0
#endif

union FragB { v16bf v; unsigned int u[8]; };

__device__ __forceinline__ unsigned short f2bf(float f){
  unsigned int x = __float_as_uint(f);
  unsigned int r = (x + 0x7FFFu + ((x >> 16) & 1u)) >> 16;
  return (unsigned short)r;
}

__device__ __forceinline__ int clampi(int v, int lo, int hi){
  return v < lo ? lo : (v > hi ? hi : v);
}

// K-offset of VGPR-pair j within a 16x32 bf16 A/B fragment (wave32 layout,
// CDNA5 ISA 7.12.2). lh = 8 for lanes 16-31, 0 for lanes 0-15.
__device__ __forceinline__ int kb_off(int j, int lh){
  return 2*j + ((j >= 4) ? 8 : 0) + lh;
}

__device__ __forceinline__ v8f wmma_bf16(v16bf a, v16bf b, v8f c){
  return __builtin_amdgcn_wmma_f32_16x16x32_bf16(false, a, false, b, (short)0, c,
                                                 false, false);
}

// ---------------------------------------------------------------------------
// LayerNorm (+ optional L2-normalized "rep" output), fp32 in -> bf16 out.
// One block per row of D=512, 256 threads (8 waves), 2 elems/thread.
// ---------------------------------------------------------------------------
template<bool WITH_REP>
__global__ __launch_bounds__(256)
void ln_kernel(const float* __restrict__ in, const float* __restrict__ g,
               const float* __restrict__ be, unsigned short* __restrict__ xo,
               unsigned short* __restrict__ ro){
  __shared__ float red[256];
  int row = blockIdx.x, tid = threadIdx.x;
  const float* p = in + (size_t)row * DD;
  float v0 = p[tid], v1 = p[tid + 256];

  red[tid] = v0 + v1; __syncthreads();
  for (int s = 128; s > 0; s >>= 1){ if (tid < s) red[tid] += red[tid + s]; __syncthreads(); }
  float mean = red[0] * (1.0f / DD); __syncthreads();

  float d0 = v0 - mean, d1 = v1 - mean;
  red[tid] = d0*d0 + d1*d1; __syncthreads();
  for (int s = 128; s > 0; s >>= 1){ if (tid < s) red[tid] += red[tid + s]; __syncthreads(); }
  float inv = rsqrtf(red[0] * (1.0f / DD) + LN_EPS); __syncthreads();

  float x0 = d0 * inv * g[tid]       + be[tid];
  float x1 = d1 * inv * g[tid + 256] + be[tid + 256];
  xo[(size_t)row*DD + tid]       = f2bf(x0);
  xo[(size_t)row*DD + tid + 256] = f2bf(x1);

  if (WITH_REP){
    red[tid] = x0*x0 + x1*x1; __syncthreads();
    for (int s = 128; s > 0; s >>= 1){ if (tid < s) red[tid] += red[tid + s]; __syncthreads(); }
    float rinv = 1.0f / fmaxf(sqrtf(red[0]), 1e-12f);
    ro[(size_t)row*DD + tid]       = f2bf(x0 * rinv);
    ro[(size_t)row*DD + tid + 256] = f2bf(x1 * rinv);
  }
}

// ---------------------------------------------------------------------------
// Tiled bf16 WMMA GEMM:  C[M,NC] = A(bf16)[M,K] @ W(f32->bf16)[K,NC] + bias
// Block 256 thr = 8 waves (2 M x 4 N), block tile 64x64, K-step 32.
// A tile staged by the Tensor Data Mover (TDM pad feature reproduces the
// 80B padded LDS row stride); W tile converted f32->bf16 by the 8 waves.
// Epilogues: QKV scatter (bf16), GELU (bf16), residual-add (fp32).
// ---------------------------------------------------------------------------
#define EPI_QKV  0
#define EPI_GELU 1
#define EPI_RES  2

template<int KDIM, int EPI>
__global__ __launch_bounds__(256)
void gemm_kernel(const unsigned short* __restrict__ A, const float* __restrict__ W,
                 const float* __restrict__ bias, const float* __restrict__ resid,
                 unsigned short* __restrict__ oq, unsigned short* __restrict__ ok2,
                 unsigned short* __restrict__ ov, unsigned short* __restrict__ obf,
                 float* __restrict__ of, int NC){
  __shared__ unsigned short As[64][40];   // [m][k], 80B row stride (16B aligned)
  __shared__ unsigned short Bs[64][40];   // [n][k] transposed for B-frag loads

  int tid = threadIdx.x, lane = tid & 31, wid = tid >> 5;
  int wm = wid & 1, wn = wid >> 1;
  int m0 = blockIdx.y * 64, n0 = blockIdx.x * 64;
  int lh = (lane & 16) ? 8 : 0;
  v8f acc0 = {}, acc1 = {};

  int lb_k = tid & 31, lb_n = (tid >> 5) * 8;    // W loader: 8 floats per thread
#if HAVE_TDM
  unsigned int lds_base = (unsigned int)(size_t)&As[0][0];
  unsigned long long gaddr0 =
      (unsigned long long)(size_t)(A + (size_t)m0 * KDIM);
#else
  int la_row = tid >> 2, la_k = (tid & 3) * 8;   // A loader: 16B per thread
#endif

  for (int k0 = 0; k0 < KDIM; k0 += 32){
    __syncthreads();
#if HAVE_TDM
    if (wid == 0){
      // D# group0: count=1 | lds_addr | global_addr[56:0] | type=2 ("image")
      unsigned long long ga = gaddr0 + (unsigned long long)k0 * 2ull;
      v4u g0;
      g0.x = 1u;
      g0.y = lds_base;
      g0.z = (unsigned int)ga;
      g0.w = ((unsigned int)(ga >> 32) & 0x01FFFFFFu) | (2u << 30);
      // D# group1: data_size=2B, pad_enable, pad every 16 DW by 4 DW (-> 80B
      // LDS row stride), tensor 512/1024 x 4096, tile 32 x 64, stride = KDIM.
      v8i g1;
      g1[0] = (int)((1u << 16) | (1u << 20) | (3u << 22) | (3u << 25));
      g1[1] = (int)((unsigned)KDIM << 16);                   // dim0[15:0]
      g1[2] = (int)(((unsigned)KDIM >> 16) |
                    (((unsigned)MM & 0xFFFFu) << 16));       // dim0 hi | dim1 lo
      g1[3] = (int)(((unsigned)MM >> 16) | (32u << 16));     // dim1 hi | tile0
      g1[4] = (int)64u;                                      // tile1=64, tile2=0
      g1[5] = (int)KDIM;                                     // dim0_stride lo
      g1[6] = 0; g1[7] = 0;
      v4i gz = {0, 0, 0, 0};
#if __clang_major__ >= 23
      v8i gz8 = {0, 0, 0, 0, 0, 0, 0, 0};
      __builtin_amdgcn_tensor_load_to_lds(g0, g1, gz, gz, gz8, 0);
#else
      __builtin_amdgcn_tensor_load_to_lds(g0, g1, gz, gz, 0);
#endif
    }
#else
    const unsigned short* ag = A + (size_t)(m0 + la_row) * KDIM + k0 + la_k;
    *(uint4*)&As[la_row][la_k] = *(const uint4*)ag;
#endif

    const float* wg = W + (size_t)(k0 + lb_k) * NC + n0 + lb_n;
    float4 w0 = *(const float4*)wg;
    float4 w1 = *(const float4*)(wg + 4);
    Bs[lb_n+0][lb_k] = f2bf(w0.x); Bs[lb_n+1][lb_k] = f2bf(w0.y);
    Bs[lb_n+2][lb_k] = f2bf(w0.z); Bs[lb_n+3][lb_k] = f2bf(w0.w);
    Bs[lb_n+4][lb_k] = f2bf(w1.x); Bs[lb_n+5][lb_k] = f2bf(w1.y);
    Bs[lb_n+6][lb_k] = f2bf(w1.z); Bs[lb_n+7][lb_k] = f2bf(w1.w);

    if (k0 + 32 < KDIM){
      __builtin_prefetch(wg + (size_t)32 * NC, 0, 1);  // global_prefetch_b8
    }
#if HAVE_TDM
    if (wid == 0) __builtin_amdgcn_s_wait_tensorcnt(0);
#endif
    __syncthreads();

    FragB a0, a1, bfr;
    #pragma unroll
    for (int j = 0; j < 8; j++){
      int kk = kb_off(j, lh);
      a0.u[j]  = *(const unsigned int*)&As[wm*32      + (lane & 15)][kk];
      a1.u[j]  = *(const unsigned int*)&As[wm*32 + 16 + (lane & 15)][kk];
      bfr.u[j] = *(const unsigned int*)&Bs[wn*16      + (lane & 15)][kk];
    }
    acc0 = wmma_bf16(a0.v, bfr.v, acc0);
    acc1 = wmma_bf16(a1.v, bfr.v, acc1);
  }

  int hi = lane >> 4;
  int col = n0 + wn*16 + (lane & 15);
  float bv = bias[col];
  #pragma unroll
  for (int s = 0; s < 2; s++){
    #pragma unroll
    for (int r = 0; r < 8; r++){
      int row = m0 + wm*32 + s*16 + r + 8*hi;    // D-layout: M = r + 8*(lane>=16)
      float val = (s ? acc1[r] : acc0[r]) + bv;
      if (EPI == EPI_QKV){
        int which = col >> 9, h = (col >> 6) & 7, hd = col & 63;
        int b = row >> 11, n = row & (NN - 1);
        size_t idx = (((size_t)(b*HH + h)) * NN + n) * HDIM + hd;
        unsigned short* dst = (which == 0) ? oq : ((which == 1) ? ok2 : ov);
        dst[idx] = f2bf(val);
      } else if (EPI == EPI_GELU){
        float gl = 0.5f * val * (1.0f + erff(val * 0.70710678118f));
        obf[(size_t)row * NC + col] = f2bf(gl);
      } else {
        of[(size_t)row * NC + col] = val + resid[(size_t)row * NC + col];
      }
    }
  }
}

// ---------------------------------------------------------------------------
// Banded representation-correlation: corr[b,qt, 16 x 160] = rep_q @ rep_k^T
// Shared by all 8 heads (8x flop saving). Block = 5 waves, one 32-key tile
// each; K loop over D=512 in steps of 32 (16 WMMAs per subtile).
// ---------------------------------------------------------------------------
__global__ __launch_bounds__(160)
void corr_kernel(const unsigned short* __restrict__ rep, float* __restrict__ corr){
  int blk = blockIdx.x;
  int b = blk >> 7, qt = blk & 127;
  int q0 = qt * 16;
  int kbase = (q0 - 64) & ~31;              // 160-wide aligned key window
  int lane = threadIdx.x & 31, w = threadIdx.x >> 5;
  int lh = (lane & 16) ? 8 : 0;
  const unsigned short* rb = rep + (size_t)b * NN * DD;
  v8f c0 = {}, c1 = {};
  int qrow = q0 + (lane & 15);
  int kg0 = kbase + w*32 + (lane & 15);
  int kc0 = clampi(kg0,      0, NN - 1);
  int kc1 = clampi(kg0 + 16, 0, NN - 1);
  for (int d0 = 0; d0 < DD; d0 += 32){
    FragB a, b0, b1;
    #pragma unroll
    for (int j = 0; j < 8; j++){
      int kk = kb_off(j, lh);
      a.u[j]  = *(const unsigned int*)&rb[(size_t)qrow * DD + d0 + kk];
      b0.u[j] = *(const unsigned int*)&rb[(size_t)kc0  * DD + d0 + kk];
      b1.u[j] = *(const unsigned int*)&rb[(size_t)kc1  * DD + d0 + kk];
    }
    c0 = wmma_bf16(a.v, b0.v, c0);
    c1 = wmma_bf16(a.v, b1.v, c1);
  }
  int hi = lane >> 4;
  float* cb = corr + (size_t)blk * 16 * 160;
  #pragma unroll
  for (int r = 0; r < 8; r++){
    int row = r + 8*hi;
    cb[(size_t)row*160 + w*32      + (lane & 15)] = c0[r];
    cb[(size_t)row*160 + w*32 + 16 + (lane & 15)] = c1[r];
  }
}

// ---------------------------------------------------------------------------
// Banded flash attention. Block = 8 waves = 8 heads, one 16-query tile.
// Per key tile of 32: q@k^T (4 WMMA) + precomputed corr + rel_bias + mask,
// online softmax (shfl_xor half-wave reductions), P@V (4 WMMA).
// Dynamic LDS: per-head K tile [32][72], V^T tile [64][40], P stage [16][40].
// ---------------------------------------------------------------------------
#define SM_KT (32*72)
#define SM_VT (64*40)
#define SM_PT (16*40)
#define ATTN_SMEM ((size_t)HH * (SM_KT + SM_VT + SM_PT) * 2)

__global__ __launch_bounds__(256)
void attn_kernel(const unsigned short* __restrict__ qb,
                 const unsigned short* __restrict__ kbuf,
                 const unsigned short* __restrict__ vbuf,
                 const float* __restrict__ corr,
                 const float* __restrict__ rel_bias,
                 const float* __restrict__ rep_scale,
                 unsigned short* __restrict__ upd){
  extern __shared__ unsigned short smem[];
  int tid = threadIdx.x, lane = tid & 31, h = tid >> 5;
  int b = blockIdx.x >> 7, qt = blockIdx.x & 127, q0 = qt * 16;
  int kbase = (q0 - 64) & ~31;
  int lh = (lane & 16) ? 8 : 0, hi = lane >> 4;

  unsigned short* Kt = smem + h * SM_KT;                       // [key][hd]
  unsigned short* Vt = smem + 8*SM_KT + h * SM_VT;             // [hd][key]
  unsigned short* Pt = smem + 8*(SM_KT + SM_VT) + h * SM_PT;   // [q][key]

  size_t hb = ((size_t)(b*HH + h)) * NN * HDIM;

  // Q fragments (16x32 bf16 x2 over HD=64), straight from global
  FragB qa0, qa1;
  int mrow = q0 + (lane & 15);
  #pragma unroll
  for (int j = 0; j < 8; j++){
    int kk = kb_off(j, lh);
    qa0.u[j] = *(const unsigned int*)&qb[hb + (size_t)mrow*HDIM      + kk];
    qa1.u[j] = *(const unsigned int*)&qb[hb + (size_t)mrow*HDIM + 32 + kk];
  }
  float rs = rep_scale[h];
  v8f O0 = {}, O1 = {}, O2 = {}, O3 = {};
  float mrun[8], lrun[8];
  #pragma unroll
  for (int r = 0; r < 8; r++){ mrun[r] = -3.0e30f; lrun[r] = 0.0f; }
  const float* cb = corr + (size_t)(b*128 + qt) * 16 * 160;

  for (int kt = 0; kt < 5; kt++){
    int kt0 = kbase + kt*32;
    int keyc = clampi(kt0 + lane, 0, NN - 1);
    const unsigned short* krow = kbuf + hb + (size_t)keyc * HDIM;
    #pragma unroll
    for (int j = 0; j < 8; j++)
      *(uint4*)&Kt[lane*72 + j*8] = *(const uint4*)&krow[j*8];
    const unsigned short* vrow = vbuf + hb + (size_t)keyc * HDIM;
    #pragma unroll
    for (int d = 0; d < 64; d++) Vt[d*40 + lane] = vrow[d];   // transpose store
    asm volatile("s_wait_dscnt 0x0" ::: "memory");

    // S = q @ k^T  (two 16-key subtiles, two 32-wide HD steps)
    v8f S0 = {}, S1 = {};
    {
      FragB bk;
      #pragma unroll
      for (int j = 0; j < 8; j++){ int kk = kb_off(j, lh);
        bk.u[j] = *(const unsigned int*)&Kt[(lane & 15)*72 + kk]; }
      S0 = wmma_bf16(qa0.v, bk.v, S0);
      #pragma unroll
      for (int j = 0; j < 8; j++){ int kk = kb_off(j, lh);
        bk.u[j] = *(const unsigned int*)&Kt[(lane & 15)*72 + 32 + kk]; }
      S0 = wmma_bf16(qa1.v, bk.v, S0);
      #pragma unroll
      for (int j = 0; j < 8; j++){ int kk = kb_off(j, lh);
        bk.u[j] = *(const unsigned int*)&Kt[(16 + (lane & 15))*72 + kk]; }
      S1 = wmma_bf16(qa0.v, bk.v, S1);
      #pragma unroll
      for (int j = 0; j < 8; j++){ int kk = kb_off(j, lh);
        bk.u[j] = *(const unsigned int*)&Kt[(16 + (lane & 15))*72 + 32 + kk]; }
      S1 = wmma_bf16(qa1.v, bk.v, S1);
    }

    // logits = S*scale + corr*rep_scale + rel_bias, band mask
    float p0[8], p1[8];
    #pragma unroll
    for (int s = 0; s < 2; s++){
      #pragma unroll
      for (int r = 0; r < 8; r++){
        int row = r + 8*hi;
        int qg = q0 + row;
        int kg = kt0 + s*16 + (lane & 15);
        int dlt = qg - kg;
        float val = -30000.0f;
        if (kg >= 0 && kg < NN && dlt <= RR && dlt >= -RR){
          float cr = cb[(size_t)row*160 + (kg - kbase)];
          float bi = rel_bias[(size_t)(dlt + NN - 1) * HH + h];
          float sv = s ? S1[r] : S0[r];
          val = sv * ASCALE + cr * rs + bi;
        }
        if (s) p1[r] = val; else p0[r] = val;
      }
    }

    // online softmax: per-row max/sum across the 16 lanes of each half-wave
    #pragma unroll
    for (int r = 0; r < 8; r++){
      float x = fmaxf(p0[r], p1[r]);
      x = fmaxf(x, __shfl_xor(x, 1, 32));
      x = fmaxf(x, __shfl_xor(x, 2, 32));
      x = fmaxf(x, __shfl_xor(x, 4, 32));
      x = fmaxf(x, __shfl_xor(x, 8, 32));
      float mn = fmaxf(mrun[r], x);
      float sc = __expf(mrun[r] - mn);
      mrun[r] = mn; lrun[r] *= sc;
      O0[r] *= sc; O1[r] *= sc; O2[r] *= sc; O3[r] *= sc;
      float e0 = __expf(p0[r] - mn); if (p0[r] <= -29999.0f) e0 = 0.0f;
      float e1 = __expf(p1[r] - mn); if (p1[r] <= -29999.0f) e1 = 0.0f;
      float sm = e0 + e1;
      sm += __shfl_xor(sm, 1, 32); sm += __shfl_xor(sm, 2, 32);
      sm += __shfl_xor(sm, 4, 32); sm += __shfl_xor(sm, 8, 32);
      lrun[r] += sm;
      int row = r + 8*hi;
      Pt[row*40      + (lane & 15)] = f2bf(e0);
      Pt[row*40 + 16 + (lane & 15)] = f2bf(e1);
    }
    asm volatile("s_wait_dscnt 0x0" ::: "memory");

    // O += P @ V   (P: 16x32 A-frag restaged via LDS; V^T B-frags)
    FragB pa;
    #pragma unroll
    for (int j = 0; j < 8; j++){ int kk = kb_off(j, lh);
      pa.u[j] = *(const unsigned int*)&Pt[(lane & 15)*40 + kk]; }
    FragB bv_;
    #pragma unroll
    for (int j = 0; j < 8; j++){ int kk = kb_off(j, lh);
      bv_.u[j] = *(const unsigned int*)&Vt[(0*16 + (lane & 15))*40 + kk]; }
    O0 = wmma_bf16(pa.v, bv_.v, O0);
    #pragma unroll
    for (int j = 0; j < 8; j++){ int kk = kb_off(j, lh);
      bv_.u[j] = *(const unsigned int*)&Vt[(1*16 + (lane & 15))*40 + kk]; }
    O1 = wmma_bf16(pa.v, bv_.v, O1);
    #pragma unroll
    for (int j = 0; j < 8; j++){ int kk = kb_off(j, lh);
      bv_.u[j] = *(const unsigned int*)&Vt[(2*16 + (lane & 15))*40 + kk]; }
    O2 = wmma_bf16(pa.v, bv_.v, O2);
    #pragma unroll
    for (int j = 0; j < 8; j++){ int kk = kb_off(j, lh);
      bv_.u[j] = *(const unsigned int*)&Vt[(3*16 + (lane & 15))*40 + kk]; }
    O3 = wmma_bf16(pa.v, bv_.v, O3);
  }

  // normalize + scatter upd[b,n, h*64+hd] as bf16
  #pragma unroll
  for (int r = 0; r < 8; r++){
    float inv = 1.0f / lrun[r];
    int row = q0 + r + 8*hi;
    size_t base = ((size_t)b * NN + row) * DD + h * HDIM;
    upd[base +  0 + (lane & 15)] = f2bf(O0[r] * inv);
    upd[base + 16 + (lane & 15)] = f2bf(O1[r] * inv);
    upd[base + 32 + (lane & 15)] = f2bf(O2[r] * inv);
    upd[base + 48 + (lane & 15)] = f2bf(O3[r] * inv);
  }
}

// ---------------------------------------------------------------------------
extern "C" void kernel_launch(void* const* d_in, const int* in_sizes, int n_in,
                              void* d_out, int out_size, void* d_ws, size_t ws_size,
                              hipStream_t stream){
  const float* nodes  = (const float*)d_in[0];
  const float* ln1_g  = (const float*)d_in[1];
  const float* ln1_b  = (const float*)d_in[2];
  const float* qkv_w  = (const float*)d_in[3];
  const float* qkv_b  = (const float*)d_in[4];
  const float* proj_w = (const float*)d_in[5];
  const float* proj_b = (const float*)d_in[6];
  const float* rep_s  = (const float*)d_in[7];
  const float* rel_b  = (const float*)d_in[8];
  const float* ln2_g  = (const float*)d_in[9];
  const float* ln2_b  = (const float*)d_in[10];
  const float* mlp_w1 = (const float*)d_in[11];
  const float* mlp_b1 = (const float*)d_in[12];
  const float* mlp_w2 = (const float*)d_in[13];
  const float* mlp_b2 = (const float*)d_in[14];

  char* ws = (char*)d_ws;
  size_t off = 0;
  auto walloc = [&](size_t bytes) -> void* {
    void* p = ws + off; off += (bytes + 255) & ~(size_t)255; return p;
  };
  unsigned short* x    = (unsigned short*)walloc((size_t)MM * DD * 2);
  unsigned short* rep  = (unsigned short*)walloc((size_t)MM * DD * 2);
  unsigned short* qbuf = (unsigned short*)walloc((size_t)MM * DD * 2);
  unsigned short* kbuf = (unsigned short*)walloc((size_t)MM * DD * 2);
  unsigned short* vbuf = (unsigned short*)walloc((size_t)MM * DD * 2);
  float*          corr = (float*)walloc((size_t)BB * 128 * 16 * 160 * 4);
  unsigned short* upd  = (unsigned short*)walloc((size_t)MM * DD * 2);
  float*          nod2 = (float*)walloc((size_t)MM * DD * 4);
  unsigned short* ybuf = (unsigned short*)walloc((size_t)MM * DD * 2);
  unsigned short* hbuf = (unsigned short*)walloc((size_t)MM * 2 * DD * 2);

  // 1) LN1 -> x(bf16), rep(bf16)
  ln_kernel<true><<<MM, 256, 0, stream>>>(nodes, ln1_g, ln1_b, x, rep);
  // 2) QKV GEMM (4096x512 @ 512x1536), scatter q/k/v [B,H,N,64] bf16
  gemm_kernel<512, EPI_QKV><<<dim3(24, 64), 256, 0, stream>>>(
      x, qkv_w, qkv_b, nullptr, qbuf, kbuf, vbuf, nullptr, nullptr, 1536);
  // 3) banded rep correlation (shared across heads)
  corr_kernel<<<BB * 128, 160, 0, stream>>>(rep, corr);
  // 4) banded flash attention -> upd bf16 [B,N,D]
  attn_kernel<<<BB * 128, 256, ATTN_SMEM, stream>>>(
      qbuf, kbuf, vbuf, corr, rel_b, rep_s, upd);
  // 5) proj GEMM + residual(nodes) -> nodes2 fp32
  gemm_kernel<512, EPI_RES><<<dim3(8, 64), 256, 0, stream>>>(
      upd, proj_w, proj_b, nodes, nullptr, nullptr, nullptr, nullptr, nod2, 512);
  // 6) LN2 -> y bf16
  ln_kernel<false><<<MM, 256, 0, stream>>>(nod2, ln2_g, ln2_b, ybuf, nullptr);
  // 7) MLP1 + exact GELU -> h bf16 [4096,1024]
  gemm_kernel<512, EPI_GELU><<<dim3(16, 64), 256, 0, stream>>>(
      ybuf, mlp_w1, mlp_b1, nullptr, nullptr, nullptr, nullptr, hbuf, nullptr, 1024);
  // 8) MLP2 + residual(nodes2) -> d_out fp32
  gemm_kernel<1024, EPI_RES><<<dim3(8, 64), 256, 0, stream>>>(
      hbuf, mlp_w2, mlp_b2, nod2, nullptr, nullptr, nullptr, nullptr,
      (float*)d_out, 512);
}
